// Net_16174846837292
// MI455X (gfx1250) — matrix operand order repaired and checked by reading, not compiled, not run on hardware
//
#include <hip/hip_runtime.h>

typedef __attribute__((ext_vector_type(8)))  _Float16 v8h;
typedef __attribute__((ext_vector_type(16))) _Float16 v16h;
typedef __attribute__((ext_vector_type(8)))  float    v8f;

#define N_NODES 4096
#define N_EDGES 8192
#define N_GRAPH 64
#define HDIM    128

__device__ __forceinline__ void atomAddF(float* p, float v) {
  __hip_atomic_fetch_add(p, v, __ATOMIC_RELAXED, __HIP_MEMORY_SCOPE_AGENT);
}

__device__ __forceinline__ v8f wmma_f16(const v16h& a, const v16h& b, const v8f& c) {
  return __builtin_amdgcn_wmma_f32_16x16x32_f16(false, a, false, b, (short)0, c, false, false);
}

// ---------------------------------------------------------------- utilities
__global__ void zero_f32(float* __restrict__ p, int n) {
  int i = blockIdx.x * blockDim.x + threadIdx.x;
  if (i < n) p[i] = 0.f;
}

// out[n*K + k] = (f16) in[k*N + n]   (convert fp32 weight [K,N] to transposed f16 [N,K])
__global__ void cvt_transpose_f16(const float* __restrict__ in, _Float16* __restrict__ out,
                                  int K, int N) {
  int i = blockIdx.x * blockDim.x + threadIdx.x;
  if (i >= K * N) return;
  int n = i / K;
  int k = i - n * K;
  out[i] = (_Float16)in[k * N + n];
}

// tiny-fan-in layer: out[r,c] = relu( bias[c] + sum_j X[r,j]*W[j,c] ), f16 output, 128 cols
__global__ void mlp_small_f16(const float* __restrict__ X, const float* __restrict__ W,
                              const float* __restrict__ B, _Float16* __restrict__ out,
                              int rows, int fan_in) {
  int i = blockIdx.x * blockDim.x + threadIdx.x;
  if (i >= rows * HDIM) return;
  int r = i >> 7, c = i & 127;
  float s = B[c];
  for (int j = 0; j < fan_in; ++j) s += X[r * fan_in + j] * W[j * HDIM + c];
  out[i] = (_Float16)fmaxf(s, 0.f);
}

// ------------------------------------------------- generic WMMA f16 GEMM
// C[M,N](f16) = act( A[M,K](f16,row-major) @ BT[N,K](f16, B transposed) + bias[N] )
// one wave per 16x16 output tile, K stepped by 64 with two independent accumulators.
// Waves in a block share tile_m -> identical A fragments served from WGP$.
__global__ __launch_bounds__(256) void wmma_gemm_f16(
    const _Float16* __restrict__ A, const _Float16* __restrict__ BT,
    const float* __restrict__ bias, _Float16* __restrict__ C,
    int M, int K, int N, int relu) {
  int wid  = (blockIdx.x * 256 + (int)threadIdx.x) >> 5;
  int lane = threadIdx.x & 31;
  int half = lane >> 4, lm = lane & 15;
  int tn = N >> 4;
  int tile_m = wid / tn, tile_n = wid - tile_m * tn;

  const _Float16* arow = A  + (size_t)(tile_m * 16 + lm) * K;
  const _Float16* bcol = BT + (size_t)(tile_n * 16 + lm) * K;

  v8f acc0 = {}, acc1 = {};
  for (int kb = 0; kb < K; kb += 64) {
#pragma unroll
    for (int u = 0; u < 2; ++u) {
      int ko = kb + 32 * u;
      // A 16x32 fragment: elem e<8 -> K = ko + 8*half + e ; elem 8+t -> K = ko + 16 + 8*half + t
      v8h alo = *(const v8h*)(arow + ko + 8 * half);
      v8h ahi = *(const v8h*)(arow + ko + 16 + 8 * half);
      // B 32x16 fragment: elem e<8 -> K = ko + 8*half + e ; elem 8+t -> K = ko + 16 + 8*half + t, column = lm (BT contiguous in K)
      v8h blo = *(const v8h*)(bcol + ko + 8 * half);
      v8h bhi = *(const v8h*)(bcol + ko + 16 + 8 * half);
      v16h a, b;
#pragma unroll
      for (int t = 0; t < 8; ++t) { a[t] = alo[t]; a[8 + t] = ahi[t]; b[t] = blo[t]; b[8 + t] = bhi[t]; }
      if (u) acc1 = wmma_f16(a, b, acc1);
      else   acc0 = wmma_f16(a, b, acc0);
    }
  }
  int col = tile_n * 16 + lm;
  float bv = bias[col];
#pragma unroll
  for (int r = 0; r < 8; ++r) {
    int row = tile_m * 16 + r + 8 * half;   // D: VGPR r -> M = r (+8 for upper half-wave)
    float v = acc0[r] + acc1[r] + bv;
    if (relu) v = fmaxf(v, 0.f);
    C[(size_t)row * N + col] = (_Float16)v;
  }
}

// ------------------------------------------------- fused e4 + per-edge matvec + scatter
// msg[e,o] = sum_{k,i} f[e,k] * h[src(e),i] * e4_w[k*16384 + i*128 + o]
//          + sum_i     h[src(e),i] * e4_b[i*128 + o]
// == (Z @ e4_w_view[16384,128]) with Z[e,(k,i)] built on the fly (outer product in registers).
// e4wT is e4_w viewed [16384,128] then transposed -> [128,16384]; e4bT is e4_b[128,128]^T.
// h fragments are register-resident; only the B stream hits memory in the loop.
// All 8 waves of a block use the SAME tile_n (same B columns, 8 different edge tiles):
// identical B addresses hit in WGP$, cutting L2 traffic ~8x.
__global__ __launch_bounds__(256) void fused_edge_msg(
    const _Float16* __restrict__ h, const _Float16* __restrict__ f,
    const _Float16* __restrict__ e4wT, const _Float16* __restrict__ e4bT,
    const int* __restrict__ src, const int* __restrict__ dst,
    float* __restrict__ agg) {
  int lane = threadIdx.x & 31;
  int half = lane >> 4, lm = lane & 15;
  int tile_n = blockIdx.x & 7;                               // block-uniform col tile
  int tile_e = (blockIdx.x >> 3) * 8 + (threadIdx.x >> 5);   // one edge tile per wave

  int erow = tile_e * 16 + lm;                         // A-row this lane feeds (dup across halves)
  const _Float16* hrow = h + (size_t)src[erow] * HDIM; // gather at load time
  const _Float16* frow = f + (size_t)erow * HDIM;
  int col = tile_n * 16 + lm;
  const _Float16* bcol  = e4wT + (size_t)col * 16384;
  const _Float16* bbcol = e4bT + col * HDIM;

  // preload this lane's h fragments for all 4 i-chunks (8 x v8h = 32 VGPRs, loop-invariant)
  v8h hfrag[8];
#pragma unroll
  for (int j = 0; j < 4; ++j) {
    hfrag[2 * j]     = *(const v8h*)(hrow + 32 * j + 8 * half);
    hfrag[2 * j + 1] = *(const v8h*)(hrow + 32 * j + 16 + 8 * half);
  }

  v8f acc0 = {}, acc1 = {};
  // main K = 16384 : outer over k (f index), inner over 4 i-chunks of 32
  for (int k = 0; k < 128; ++k) {
    _Float16 fk = frow[k];
    const _Float16* bk = bcol + k * 128 + 8 * half;
    __builtin_prefetch(bcol + (k + 8) * 128, 0, 1);    // pull B stream ahead (global_prefetch_b8)
#pragma unroll
    for (int j = 0; j < 4; ++j) {
      v8h blo = *(const v8h*)(bk + 32 * j);
      v8h bhi = *(const v8h*)(bk + 32 * j + 16);
      v16h a, b;
#pragma unroll
      for (int t = 0; t < 8; ++t) {
        a[t] = fk * hfrag[2 * j][t]; a[8 + t] = fk * hfrag[2 * j + 1][t]; // Z tile in-register
        b[t] = blo[t];               b[8 + t] = bhi[t];
      }
      if (j & 1) acc1 = wmma_f16(a, b, acc1);
      else       acc0 = wmma_f16(a, b, acc0);
    }
  }
  // bias path: extra K = 128 with A = h, B = e4_b viewed [128,128]
#pragma unroll
  for (int j = 0; j < 4; ++j) {
    v8h blo = *(const v8h*)(bbcol + 32 * j + 8 * half);
    v8h bhi = *(const v8h*)(bbcol + 32 * j + 16 + 8 * half);
    v16h a, b;
#pragma unroll
    for (int t = 0; t < 8; ++t) {
      a[t] = hfrag[2 * j][t]; a[8 + t] = hfrag[2 * j + 1][t];
      b[t] = blo[t];          b[8 + t] = bhi[t];
    }
    if (j & 1) acc1 = wmma_f16(a, b, acc1);
    else       acc0 = wmma_f16(a, b, acc0);
  }
  // scatter-add msg rows to destination nodes
#pragma unroll
  for (int r = 0; r < 8; ++r) {
    int e = tile_e * 16 + r + 8 * half;
    atomAddF(agg + (size_t)dst[e] * HDIM + col, acc0[r] + acc1[r]);
  }
}

// ------------------------------------------------- scatter-mean + global add pool
__global__ void count_deg(const int* __restrict__ dst, float* __restrict__ cnt) {
  int e = blockIdx.x * blockDim.x + threadIdx.x;
  if (e < N_EDGES) atomAddF(cnt + dst[e], 1.f);
}

__global__ void mean_pool(const float* __restrict__ agg, const float* __restrict__ cnt,
                          const int* __restrict__ batch, float* __restrict__ out) {
  int i = blockIdx.x * blockDim.x + threadIdx.x;
  if (i >= N_NODES * HDIM) return;
  int n = i >> 7, c = i & 127;
  float v = agg[i] / fmaxf(cnt[n], 1.f);
  atomAddF(out + (size_t)batch[n] * HDIM + c, v);
}

// ---------------------------------------------------------------- launch
extern "C" void kernel_launch(void* const* d_in, const int* in_sizes, int n_in,
                              void* d_out, int out_size, void* d_ws, size_t ws_size,
                              hipStream_t stream) {
  (void)in_sizes; (void)n_in; (void)out_size; (void)ws_size;
  const float* x         = (const float*)d_in[0];
  const float* edge_attr = (const float*)d_in[1];
  const int*   edge_idx  = (const int*)d_in[2];
  const int*   batch     = (const int*)d_in[3];
  const float* p1_w = (const float*)d_in[4];  const float* p1_b = (const float*)d_in[5];
  const float* p2_w = (const float*)d_in[6];  const float* p2_b = (const float*)d_in[7];
  const float* e1_w = (const float*)d_in[8];  const float* e1_b = (const float*)d_in[9];
  const float* e2_w = (const float*)d_in[10]; const float* e2_b = (const float*)d_in[11];
  const float* e3_w = (const float*)d_in[12]; const float* e3_b = (const float*)d_in[13];
  const float* e4_w = (const float*)d_in[14]; const float* e4_b = (const float*)d_in[15];
  const int* src = edge_idx;
  const int* dst = edge_idx + N_EDGES;
  float* out = (float*)d_out;

  char* ws = (char*)d_ws;
  _Float16* h1   = (_Float16*)(ws + 0);          // 4096*128 f16 = 1 MB
  _Float16* hf   = (_Float16*)(ws + 1048576);    // 4096*128 f16 = 1 MB
  _Float16* e1f  = (_Float16*)(ws + 2097152);    // 8192*128 f16 = 2 MB
  _Float16* e2f  = (_Float16*)(ws + 4194304);    // 8192*256 f16 = 4 MB
  _Float16* ff   = (_Float16*)(ws + 8388608);    // 8192*128 f16 = 2 MB
  _Float16* p2wT = (_Float16*)(ws + 10485760);   // 128*128
  _Float16* e2wT = (_Float16*)(ws + 10518528);   // 256*128
  _Float16* e3wT = (_Float16*)(ws + 10584064);   // 128*256
  _Float16* e4wT = (_Float16*)(ws + 10649600);   // 128*16384 f16 = 4 MB
  _Float16* e4bT = (_Float16*)(ws + 14843904);   // 128*128
  float*    agg  = (float*)   (ws + 14876672);   // 4096*128 f32 = 2 MB
  float*    cnt  = (float*)   (ws + 16973824);   // 4096 f32

  // zero accumulators + output (harness poisons them)
  zero_f32<<<(N_NODES * HDIM + 255) / 256, 256, 0, stream>>>(agg, N_NODES * HDIM);
  zero_f32<<<(N_NODES + 255) / 256, 256, 0, stream>>>(cnt, N_NODES);
  zero_f32<<<(N_GRAPH * HDIM + 255) / 256, 256, 0, stream>>>(out, N_GRAPH * HDIM);

  // f16 transposed weight copies (L2-resident; e4_w f16 is 4 MB, reused 512x)
  cvt_transpose_f16<<<(128 * 128 + 255) / 256, 256, 0, stream>>>(p2_w, p2wT, 128, 128);
  cvt_transpose_f16<<<(128 * 256 + 255) / 256, 256, 0, stream>>>(e2_w, e2wT, 128, 256);
  cvt_transpose_f16<<<(256 * 128 + 255) / 256, 256, 0, stream>>>(e3_w, e3wT, 256, 128);
  cvt_transpose_f16<<<(16384 * 128 + 255) / 256, 256, 0, stream>>>(e4_w, e4wT, 16384, 128);
  cvt_transpose_f16<<<(128 * 128 + 255) / 256, 256, 0, stream>>>(e4_b, e4bT, 128, 128);

  // node preprocess MLP: 11 -> 128 (VALU, tiny) -> 128 (WMMA)
  mlp_small_f16<<<(N_NODES * HDIM + 255) / 256, 256, 0, stream>>>(x, p1_w, p1_b, h1, N_NODES, 11);
  wmma_gemm_f16<<<(4096 / 16) * (128 / 16) / 8, 256, 0, stream>>>(h1, p2wT, p2_b, hf, 4096, 128, 128, 0);

  // edge MLP: 5 -> 128 (VALU, tiny) -> 256 -> 128 (WMMA)
  mlp_small_f16<<<(N_EDGES * HDIM + 255) / 256, 256, 0, stream>>>(edge_attr, e1_w, e1_b, e1f, N_EDGES, 5);
  wmma_gemm_f16<<<(8192 / 16) * (256 / 16) / 8, 256, 0, stream>>>(e1f, e2wT, e2_b, e2f, 8192, 128, 256, 1);
  wmma_gemm_f16<<<(8192 / 16) * (128 / 16) / 8, 256, 0, stream>>>(e2f, e3wT, e3_b, ff, 8192, 256, 128, 1);

  // degree counts for mean aggregation
  count_deg<<<(N_EDGES + 255) / 256, 256, 0, stream>>>(dst, cnt);

  // fused e4-GEMM + per-edge matvec + scatter-add (34.4 GF, never materializes W)
  fused_edge_msg<<<512, 256, 0, stream>>>(hf, ff, e4wT, e4bT, src, dst, agg);

  // mean + global add pool
  mean_pool<<<(N_NODES * HDIM + 255) / 256, 256, 0, stream>>>(agg, cnt, batch, out);
}
